// GlobalTokenMergeModule_76845554860102
// MI455X (gfx1250) — matrix operand order repaired and checked by reading, not compiled, not run on hardware
//
#include <hip/hip_runtime.h>
#include <hip/hip_bf16.h>

// Problem constants (from setup_inputs): B=2, L=2048, D=512, N=4096, G=64, target=512
#define BB    2
#define LL    2048
#define DD    512
#define NN    4096
#define GG    64
#define TGT   512
#define LPAD  2048
#define MAX_ROUNDS 24

typedef __attribute__((ext_vector_type(16))) __bf16 v16bf;
typedef __attribute__((ext_vector_type(8)))  float  v8f;

// ---------------------------------------------------------------- f32 -> bf16
__global__ void cvt_bf16_kernel(const float* __restrict__ in,
                                __bf16* __restrict__ out, long n) {
  long i = (long)blockIdx.x * blockDim.x + threadIdx.x;
  if (i < n) out[i] = (__bf16)in[i];
}

// ------------------------------------------------- g = x @ w^T  via WMMA bf16
// One wave per 16x16 output tile.  M = B*L rows, N = G cols, K = D.
__global__ __launch_bounds__(256)
void proj_wmma_kernel(const __bf16* __restrict__ xbf,   // (B*L, D) row major
                      const __bf16* __restrict__ wbf,   // (G, D)   row major
                      float* __restrict__ g)            // (B*L, G)
{
  const int wave = (blockIdx.x * blockDim.x + threadIdx.x) >> 5;
  const int lane = threadIdx.x & 31;
  const int nTilesN = GG >> 4;                 // 4
  const int row_tile = wave / nTilesN;
  const int col_tile = wave % nTilesN;
  if (row_tile * 16 >= BB * LL) return;

  const int m    = lane & 15;
  const int half = lane >> 4;
  const int arow = row_tile * 16 + m;          // A row this lane holds
  const int bcol = col_tile * 16 + (lane & 15);// B column (w_group row)

  v8f acc = {};
  for (int kb = 0; kb < DD; kb += 32) {
    v16bf a, bmat;
    // A 16x32 bf16 layout: VGPR v, lanes 0-15: K=2v(+1) (v<4) / 16+2(v-4)
    //                      lanes 16-31: +8 within each half     (ISA 7.12.2)
    #pragma unroll
    for (int v = 0; v < 8; ++v) {
      const int ka = ((v < 4) ? 0 : 16) + half * 8 + 2 * (v & 3);
      a[2 * v]     = xbf[(size_t)arow * DD + kb + ka];
      a[2 * v + 1] = xbf[(size_t)arow * DD + kb + ka + 1];
      // B 32x16 bf16 layout: same K mapping as A
      const int kw = ((v < 4) ? 0 : 16) + half * 8 + 2 * (v & 3);
      bmat[2 * v]     = wbf[(size_t)bcol * DD + kb + kw];
      bmat[2 * v + 1] = wbf[(size_t)bcol * DD + kb + kw + 1];
    }
    acc = __builtin_amdgcn_wmma_f32_16x16x32_bf16(
        /*neg_a=*/false, a, /*neg_b=*/false, bmat,
        /*c_mod=*/(short)0, acc, /*reuse_a=*/false, /*reuse_b=*/false);
  }
  // C/D layout: VGPR v, lanes 0-15 -> M=v, lanes 16-31 -> M=8+v; N=lane%16
  #pragma unroll
  for (int v = 0; v < 8; ++v) {
    const int mm = v + 8 * half;
    g[(size_t)(row_tile * 16 + mm) * GG + col_tile * 16 + (lane & 15)] = acc[v];
  }
}

// ---------------------------------------- norms + normalized group vectors gn
__global__ __launch_bounds__(256)
void norm_kernel(const float* __restrict__ g,
                 float* __restrict__ gn, float* __restrict__ norms, int rows) {
  const int row  = (blockIdx.x * blockDim.x + threadIdx.x) >> 5;
  const int lane = threadIdx.x & 31;
  if (row >= rows) return;
  const float v0 = g[(size_t)row * GG + lane];
  const float v1 = g[(size_t)row * GG + lane + 32];
  float ss = v0 * v0 + v1 * v1;
  #pragma unroll
  for (int off = 16; off > 0; off >>= 1) ss += __shfl_xor(ss, off, 32);
  const float nrm = sqrtf(ss);
  const float inv = 1.0f / fmaxf(nrm, 1e-12f);
  gn[(size_t)row * GG + lane]      = v0 * inv;
  gn[(size_t)row * GG + lane + 32] = v1 * inv;
  if (lane == 0) norms[row] = nrm;
}

// ----------------------------------------------------- merge simulator (1 WG)
// Clusters always span contiguous original-row ranges (adjacent merges only),
// so we only track gn (global, ping-pong), norms + [begin,end) ranges (LDS)
// and a per-original-row scalar coefficient for the weighted x average.
__global__ __launch_bounds__(1024)
void merge_kernel(float* __restrict__ gn0, float* __restrict__ gn1,
                  const float* __restrict__ norms_g,
                  float* __restrict__ coeff, int* __restrict__ ranges) {
  __shared__ float key[LPAD];
  __shared__ int   sidx[LPAD];
  __shared__ float normsA[LPAD], normsB[LPAD];
  __shared__ unsigned short begA[LPAD], endA[LPAD], begB[LPAD], endB[LPAD];
  __shared__ unsigned char  msk[LPAD], pst[LPAD];
  __shared__ int   scan[LPAD];
  __shared__ int sh_n, sh_rem, sh_cnt, sh_par;

  const int b = blockIdx.x;
  const int tid = threadIdx.x;
  const int lane = tid & 31, wid = tid >> 5, nw = blockDim.x >> 5;

  for (int i = tid; i < LL; i += blockDim.x) {
    normsA[i] = norms_g[b * LL + i];
    begA[i] = (unsigned short)i;
    endA[i] = (unsigned short)(i + 1);
    coeff[b * LL + i] = 1.0f;
  }
  if (tid == 0) { sh_n = LL; sh_rem = LL - TGT; sh_par = 0; }
  __syncthreads();

  for (int round = 0; round < MAX_ROUNDS; ++round) {
    const int n = sh_n, rem = sh_rem, par = sh_par;
    if (rem <= 0 || n < 2) break;
    float* gsrc = par ? gn1 : gn0;
    float* gdst = par ? gn0 : gn1;
    float*          nsrc = par ? normsB : normsA;
    float*          ndst = par ? normsA : normsB;
    unsigned short* bsrc = par ? begB : begA;
    unsigned short* bdst = par ? begA : begB;
    unsigned short* esrc = par ? endB : endA;
    unsigned short* edst = par ? endA : endB;

    // 1) adjacent similarities (dot of unit vectors), padded with -inf
    for (int i = tid; i < LPAD; i += blockDim.x) {
      float s = -3.0e38f;
      if (i < n - 1) {
        const float* r0 = gsrc + ((size_t)b * LL + i) * GG;
        const float* r1 = r0 + GG;
        float acc = 0.0f;
        #pragma unroll 8
        for (int d = 0; d < GG; ++d) acc += r0[d] * r1[d];
        s = acc;
      }
      key[i] = s; sidx[i] = i;
      msk[i] = 0; pst[i] = 0;
    }
    __syncthreads();

    // 2) bitonic sort, descending by key (argsort(-sim))
    for (int k = 2; k <= LPAD; k <<= 1) {
      for (int j = k >> 1; j > 0; j >>= 1) {
        for (int t = tid; t < LPAD; t += blockDim.x) {
          const int ixj = t ^ j;
          if (ixj > t) {
            const float a = key[t], c = key[ixj];
            const bool up = ((t & k) == 0);
            if (up ? (a < c) : (a > c)) {
              key[t] = c; key[ixj] = a;
              const int ti = sidx[t]; sidx[t] = sidx[ixj]; sidx[ixj] = ti;
            }
          }
        }
        __syncthreads();
      }
    }

    // 3) serial greedy pair selection (thread 0, matches lax.scan semantics)
    if (tid == 0) {
      const int r_step = min(rem, n >> 1);
      int cnt = 0;
      for (int t = 0; t < LPAD && cnt < r_step; ++t) {
        const int i = sidx[t];
        if (i < n - 1 && !msk[i] && !msk[i + 1]) {
          msk[i] = 1; msk[i + 1] = 1; pst[i] = 1; ++cnt;
        }
      }
      sh_cnt = cnt;
    }
    __syncthreads();

    // 4) keep flags + inclusive prefix scan -> compaction destinations
    for (int i = tid; i < LPAD; i += blockDim.x) {
      const int second = (i > 0) && pst[i - 1];
      scan[i] = (i < n && !second) ? 1 : 0;
    }
    __syncthreads();
    for (int off = 1; off < LPAD; off <<= 1) {
      const int i0 = tid, i1 = tid + 1024;
      const int v0 = (i0 >= off) ? scan[i0 - off] : 0;
      const int v1 = (i1 >= off) ? scan[i1 - off] : 0;
      __syncthreads();
      scan[i0] += v0; scan[i1] += v1;
      __syncthreads();
    }

    // 5) merge + compact (one wave per kept cluster; 32 lanes cover 64 dims)
    for (int i = wid; i < n; i += nw) {
      const int second = (i > 0) && pst[i - 1];
      if (second) continue;
      const int d = scan[i] - 1;
      const float* s0 = gsrc + ((size_t)b * LL + i) * GG;
      float*       dp = gdst + ((size_t)b * LL + d) * GG;
      if (pst[i]) {
        const float wi = nsrc[i], wj = nsrc[i + 1];
        const float tot = wi + wj + 1e-8f;
        const float* s1 = s0 + GG;
        float v0 = (wi * s0[lane]      + wj * s1[lane])      / tot;
        float v1 = (wi * s0[lane + 32] + wj * s1[lane + 32]) / tot;
        float ss = v0 * v0 + v1 * v1;
        #pragma unroll
        for (int off = 16; off > 0; off >>= 1) ss += __shfl_xor(ss, off, 32);
        const float inv = 1.0f / fmaxf(sqrtf(ss), 1e-12f);
        dp[lane] = v0 * inv; dp[lane + 32] = v1 * inv;
        if (lane == 0) {
          ndst[d] = 0.5f * (wi + wj);
          bdst[d] = bsrc[i];
          edst[d] = esrc[i + 1];
        }
        // propagate x-merge weights to original rows (disjoint ranges -> no race)
        const float fi = wi / tot, fj = wj / tot;
        for (int r = bsrc[i] + lane;     r < (int)esrc[i];     r += 32) coeff[b * LL + r] *= fi;
        for (int r = bsrc[i + 1] + lane; r < (int)esrc[i + 1]; r += 32) coeff[b * LL + r] *= fj;
      } else {
        dp[lane] = s0[lane]; dp[lane + 32] = s0[lane + 32];
        if (lane == 0) { ndst[d] = nsrc[i]; bdst[d] = bsrc[i]; edst[d] = esrc[i]; }
      }
    }
    __syncthreads();
    if (tid == 0) { sh_n = n - sh_cnt; sh_rem = rem - sh_cnt; sh_par = par ^ 1; }
    __syncthreads();
  }

  // emit final cluster ranges (first TGT clusters, like x[:target_len])
  const int fpar = sh_par;
  unsigned short* bfin = fpar ? begB : begA;
  unsigned short* efin = fpar ? endB : endA;
  for (int c = tid; c < TGT; c += blockDim.x) {
    ranges[(b * TGT + c) * 2]     = bfin[c];
    ranges[(b * TGT + c) * 2 + 1] = efin[c];
  }
}

// ---------------------------------------------- x output: weighted range sum
__global__ __launch_bounds__(256)
void out_x_kernel(const float* __restrict__ x, const float* __restrict__ coeff,
                  const int* __restrict__ ranges, float* __restrict__ outx) {
  const int c = blockIdx.x;            // 0 .. B*TGT-1
  const int b = c / TGT;
  const int beg = ranges[c * 2], end = ranges[c * 2 + 1];
  const int d0 = threadIdx.x, d1 = threadIdx.x + 256;
  float a0 = 0.0f, a1 = 0.0f;
  for (int r = beg; r < end; ++r) {
    const float w = coeff[b * LL + r];
    const float* row = x + ((size_t)b * LL + r) * DD;
    a0 += w * row[d0];
    a1 += w * row[d1];
  }
  float* o = outx + (size_t)c * DD;
  o[d0] = a0; o[d1] = a1;
}

// ------------------------------------------------- s output: plain range sum
__global__ __launch_bounds__(1024)
void out_s_kernel(const float* __restrict__ s, const int* __restrict__ ranges,
                  float* __restrict__ outs) {
  const int c = blockIdx.x;            // 0 .. B*TGT-1
  const int b = c / TGT;
  const int beg = ranges[c * 2], end = ranges[c * 2 + 1];
  const int col = threadIdx.x * 4;     // 1024 threads * float4 == 4096 cols
  float4 acc = {0.f, 0.f, 0.f, 0.f};
  for (int r = beg; r < end; ++r) {
    const float4 v = *(const float4*)(s + ((size_t)b * LL + r) * NN + col);
    acc.x += v.x; acc.y += v.y; acc.z += v.z; acc.w += v.w;
  }
  *(float4*)(outs + (size_t)c * NN + col) = acc;
}

// ---------------------------------------------------------------------------
extern "C" void kernel_launch(void* const* d_in, const int* in_sizes, int n_in,
                              void* d_out, int out_size, void* d_ws, size_t ws_size,
                              hipStream_t stream) {
  const float* x  = (const float*)d_in[0];   // (B, L, D)
  const float* s  = (const float*)d_in[1];   // (B, L, N)
  const float* w  = (const float*)d_in[2];   // (G, D)

  // workspace carve-up (all 256B aligned)
  char* ws = (char*)d_ws;
  size_t off = 0;
  auto alloc = [&](size_t bytes) {
    void* p = ws + off;
    off += (bytes + 255) & ~(size_t)255;
    return p;
  };
  __bf16* xbf   = (__bf16*)alloc((size_t)BB * LL * DD * sizeof(__bf16));
  __bf16* wbf   = (__bf16*)alloc((size_t)GG * DD * sizeof(__bf16));
  float*  g     = (float*) alloc((size_t)BB * LL * GG * sizeof(float));
  float*  gn0   = (float*) alloc((size_t)BB * LL * GG * sizeof(float));
  float*  gn1   = (float*) alloc((size_t)BB * LL * GG * sizeof(float));
  float*  norms = (float*) alloc((size_t)BB * LL * sizeof(float));
  float*  coeff = (float*) alloc((size_t)BB * LL * sizeof(float));
  int*    ranges= (int*)   alloc((size_t)BB * TGT * 2 * sizeof(int));
  (void)ws_size; (void)in_sizes; (void)n_in; (void)out_size;

  float* outx = (float*)d_out;                       // (B, TGT, D)
  float* outs = (float*)d_out + (size_t)BB * TGT * DD; // (B, TGT, N)

  // 1) bf16 conversions
  {
    long nx = (long)BB * LL * DD;
    cvt_bf16_kernel<<<(unsigned)((nx + 255) / 256), 256, 0, stream>>>(x, xbf, nx);
    long nw = (long)GG * DD;
    cvt_bf16_kernel<<<(unsigned)((nw + 255) / 256), 256, 0, stream>>>(w, wbf, nw);
  }
  // 2) WMMA projection: 1024 tiles, 8 waves per 256-thread block
  proj_wmma_kernel<<<(BB * LL / 16) * (GG / 16) / 8, 256, 0, stream>>>(xbf, wbf, g);
  // 3) norms + gn (one wave per row)
  norm_kernel<<<(BB * LL) / 8, 256, 0, stream>>>(g, gn0, norms, BB * LL);
  // 4) greedy merge simulation, one workgroup per batch
  merge_kernel<<<BB, 1024, 0, stream>>>(gn0, gn1, norms, coeff, ranges);
  // 5) outputs
  out_x_kernel<<<BB * TGT, 256, 0, stream>>>(x, coeff, ranges, outx);
  out_s_kernel<<<BB * TGT, 1024, 0, stream>>>(s, ranges, outs);
}
